// RNN_50_100_76218489634995
// MI455X (gfx1250) — matrix operand order repaired and checked by reading, not compiled, not run on hardware
//
#include <hip/hip_runtime.h>
#include <hip/hip_bf16.h>
#include <stdint.h>

#define IN_DIM 50
#define NY     100
#define HID    256
#define GATES  1024
#define STEPS  100
#define BATCH  16384

// padded LDS strides (elements) to avoid bank conflicts
#define S_W  1032   // weight chunk row stride (1024 + 8)
#define S_H  264    // ht row stride (256 + 8)
#define S_WO 120    // Wo row stride (112 + 8)

typedef __attribute__((ext_vector_type(16))) __bf16 v16bf;
typedef __attribute__((ext_vector_type(8)))  float  v8f;
typedef __attribute__((ext_vector_type(4)))  float  v4f;

__device__ __forceinline__ float fast_sigmoid(float x) {
    return 1.0f / (1.0f + __expf(-x));
}
__device__ __forceinline__ float fast_tanh(float x) {
    return 1.0f - 2.0f / (__expf(2.0f * x) + 1.0f);
}
__device__ __forceinline__ v8f splat8(float x) {
    v8f v;
#pragma unroll
    for (int i = 0; i < 8; i++) v[i] = x;
    return v;
}
// Build a 16xbf16 WMMA fragment from two 16B LDS reads (ds_load_b128 x2)
__device__ __forceinline__ v16bf ld_frag(const __bf16* p0, const __bf16* p1) {
    union { v4f f[2]; v16bf v; } u;
    u.f[0] = *(const v4f*)p0;
    u.f[1] = *(const v4f*)p1;
    return u.v;
}

// ---- CDNA5 async global->LDS copy (ASYNCcnt tracked) ----
__device__ __forceinline__ void async_ld16(uint32_t lds_off, const void* gaddr) {
    asm volatile("global_load_async_to_lds_b128 %0, %1, off"
                 :: "v"(lds_off), "v"(gaddr) : "memory");
}
__device__ __forceinline__ void wait_async8() {
    asm volatile("s_wait_asynccnt 8" ::: "memory");
}
__device__ __forceinline__ void wait_async0() {
    asm volatile("s_wait_asynccnt 0" ::: "memory");
}

// Issue one 64KB weight K-chunk (32 K-rows x 1024 gate cols, bf16) into LDS.
// 4096 16B vectors / 512 threads = 8 async loads per thread (8 per wave on ASYNCcnt).
__device__ __forceinline__ void issue_chunk(const __bf16* WgB, int kc,
                                            uint32_t lds_base, int tid) {
    const char* gbase = (const char*)WgB + (size_t)kc * 32 * 1024 * 2;
#pragma unroll
    for (int j = 0; j < 8; j++) {
        int v = tid + j * 512;          // 0..4095
        int row = v >> 7;               // 0..31
        int c   = v & 127;              // 16B column within row
        uint32_t loff = lds_base + (uint32_t)(row * (S_W * 2) + c * 16);
        const void* g = gbase + row * 2048 + c * 16;
        async_ld16(loff, g);
    }
}

// -------- prep: fold Wih+Whh into bf16 B-layout [K=256][N=1024]; Wo -> [K=256][N=112] --------
__global__ void prep_gates(const float* __restrict__ Wih, const float* __restrict__ Whh,
                           const float* __restrict__ bih, const float* __restrict__ bhh,
                           const float* __restrict__ Wo,
                           __bf16* __restrict__ WgB, float* __restrict__ bg,
                           __bf16* __restrict__ WoB) {
    int idx = blockIdx.x * 256 + threadIdx.x;
    if (idx < HID * GATES) {
        int k = idx >> 10, n = idx & 1023;
        WgB[idx] = (__bf16)(Wih[n * HID + k] + Whh[n * HID + k]);
    }
    if (idx < GATES) bg[idx] = bih[idx] + bhh[idx];
    if (idx < HID * 112) {
        int k = idx / 112, n = idx - k * 112;
        WoB[idx] = (n < NY) ? (__bf16)Wo[n * HID + k] : (__bf16)0.0f;
    }
}

// -------- tiny MLP layers (f32; <0.3% of total FLOPs) --------
template <bool BF16OUT>
__global__ void mlp_layer(const float* __restrict__ X, const float* __restrict__ W,
                          const float* __restrict__ b, void* __restrict__ Y,
                          int K, int N) {
    int tid = blockIdx.x * blockDim.x + threadIdx.x;
    int r = tid / N, n = tid - r * N;
    const float* x = X + (size_t)r * K;
    const float* w = W + (size_t)n * K;
    float acc = b[n];
    for (int k = 0; k < K; k++) acc = fmaf(x[k], w[k], acc);
    acc = fmaxf(acc, 0.0f);
    if (BF16OUT) ((__bf16*)Y)[tid] = (__bf16)acc;
    else         ((float*)Y)[tid]  = acc;
}

// -------- main LSTM kernel: 512 threads (16 waves), 32 batch rows per block --------
__global__ __launch_bounds__(512, 1)
void lstm_kernel(const __bf16* __restrict__ h0, const __bf16* __restrict__ WgB,
                 const float* __restrict__ bg, const __bf16* __restrict__ WoB,
                 const float* __restrict__ bo, float* __restrict__ out) {
    __shared__ alignas(16) __bf16 lds_w[2][32 * S_W];   // 129 KB  double-buffered weight chunks
    __shared__ alignas(16) __bf16 lds_h[2][32 * S_H];   // 33 KB   ht double buffer
    __shared__ alignas(16) __bf16 lds_wo[256 * S_WO];   // 60 KB   resident Wo^T

    const int tid  = threadIdx.x;
    const int w    = tid >> 5;
    const int lane = tid & 31;
    const int m    = lane & 15;     // column within 16-wide tile / row within A tile
    const int half = lane >> 4;     // K-half selector for A, row-half for C
    const int rg   = w >> 3;        // row group 0..1  -> rows [16*rg, 16*rg+16)
    const int cg   = w & 7;         // col group 0..7  -> hidden col tiles {2cg, 2cg+1}
    const int rbase  = rg * 16;
    const int n0     = cg * 2;
    const int rstart = blockIdx.x * 32;

    // stage Wo (once)
    {
        const uint4* src = (const uint4*)WoB;       // 3584 x 16B
#pragma unroll
        for (int j = 0; j < 7; j++) {
            int v = tid + j * 512;
            int row = v / 14, c = v - row * 14;
            *(uint4*)&lds_wo[row * S_WO + c * 8] = src[v];
        }
    }
    // stage h0 -> lds_h[0]
    {
        const uint4* src = (const uint4*)(h0 + (size_t)rstart * HID);  // 1024 x 16B
#pragma unroll
        for (int j = 0; j < 2; j++) {
            int v = tid + j * 512;
            int row = v >> 5, c = v & 31;
            *(uint4*)&lds_h[0][row * S_H + c * 8] = src[v];
        }
    }

    // per-lane gate biases and c0 (accumulator layout: elem r -> row rbase+r+8*half, col n*16+m)
    float bias[2][4];
    v8f ct[2];
#pragma unroll
    for (int t = 0; t < 2; t++) {
        int n = n0 + t;
#pragma unroll
        for (int g = 0; g < 4; g++) bias[t][g] = bg[g * HID + n * 16 + m];
#pragma unroll
        for (int r = 0; r < 8; r++) {
            int row = rstart + rbase + r + 8 * half;
            ct[t][r] = (float)h0[(size_t)row * HID + n * 16 + m];
        }
    }

    // y-projection tile assignment: 2 row tiles x 7 col tiles = 14 tiles for waves 0..13
    const bool has_y = (w < 14);
    int yrt = 0, nt16 = 0, ycol = 0;
    float yb = 0.0f;
    if (has_y) {
        yrt  = w / 7;
        nt16 = (w - yrt * 7) * 16;
        ycol = nt16 + m;
        yb   = (ycol < NY) ? bo[ycol] : 0.0f;
    }

    __syncthreads();

    const uint32_t ldsw_base0 = (uint32_t)(uintptr_t)&lds_w[0][0];
    const uint32_t ldsw_base1 = (uint32_t)(uintptr_t)&lds_w[1][0];

    for (int s = 0; s < STEPS; s++) {
        const __bf16* hcur  = lds_h[s & 1];
        __bf16*       hnext = lds_h[(s + 1) & 1];

        // -------- Phase A: gates = ht @ Wg + bg  (K=256, bf16 WMMA, async dbl-buffered weights)
        v8f acc[2][4];
#pragma unroll
        for (int t = 0; t < 2; t++)
#pragma unroll
            for (int g = 0; g < 4; g++) acc[t][g] = splat8(bias[t][g]);

        issue_chunk(WgB, 0, ldsw_base0, tid);
#pragma unroll
        for (int kc = 0; kc < 8; kc++) {
            if (kc < 7) {
                issue_chunk(WgB, kc + 1, (kc & 1) ? ldsw_base0 : ldsw_base1, tid);
                wait_async8();              // current chunk landed; next still in flight
            } else {
                wait_async0();
            }
            __syncthreads();                // chunk visible to all waves

            const __bf16* wbuf = lds_w[kc & 1];
            const int k0 = kc * 32 + half * 8;
            const __bf16* hrow = hcur + (rbase + m) * S_H;
            v16bf a = ld_frag(hrow + k0, hrow + k0 + 16);

            const __bf16* wrow = wbuf + m;   // lane = N column within tile; K varies over elements
#pragma unroll
            for (int t = 0; t < 2; t++) {
                const int cbase = (n0 + t) * 16;
#pragma unroll
                for (int g = 0; g < 4; g++) {
                    v16bf b;
#pragma unroll
                    for (int e = 0; e < 16; e++) {
                        const int kk = (e < 8) ? (half * 8 + e) : (16 + half * 8 + (e - 8));
                        b[e] = wrow[kk * S_W + g * HID + cbase];
                    }
                    acc[t][g] = __builtin_amdgcn_wmma_f32_16x16x32_bf16(
                        false, a, false, b, (short)0, acc[t][g], false, false);
                }
            }
            __syncthreads();                // all waves done before buffer reuse
        }

        // -------- Phase B: elementwise LSTM cell update; write ht(bf16) to next buffer
#pragma unroll
        for (int t = 0; t < 2; t++) {
            const int n = n0 + t;
#pragma unroll
            for (int r = 0; r < 8; r++) {
                float ig = fast_sigmoid(acc[t][0][r]);
                float fg = fast_sigmoid(acc[t][1][r]);
                float gg = fast_tanh   (acc[t][2][r]);
                float og = fast_sigmoid(acc[t][3][r]);
                float c  = fg * ct[t][r] + ig * gg;
                ct[t][r] = c;
                float h  = og * fast_tanh(c);
                hnext[(rbase + r + 8 * half) * S_H + n * 16 + m] = (__bf16)h;
            }
        }
        __syncthreads();                    // new ht visible

        // -------- Phase C: y = ht_new @ Wo^T + bo, stream to HBM
        if (has_y) {
            v8f ya = splat8(yb);
            const int rb2 = yrt * 16;
#pragma unroll
            for (int kc = 0; kc < 8; kc++) {
                const int k0 = kc * 32 + half * 8;
                const __bf16* hrow = hnext + (rb2 + m) * S_H;
                v16bf a = ld_frag(hrow + k0, hrow + k0 + 16);
                const __bf16* wrow = lds_wo + (kc * 32) * S_WO + nt16 + m;
                v16bf b;
#pragma unroll
                for (int e = 0; e < 16; e++) {
                    const int kk = (e < 8) ? (half * 8 + e) : (16 + half * 8 + (e - 8));
                    b[e] = wrow[kk * S_WO];
                }
                ya = __builtin_amdgcn_wmma_f32_16x16x32_bf16(
                    false, a, false, b, (short)0, ya, false, false);
            }
            if (ycol < NY) {
                const size_t rowb = (size_t)(rstart + rb2 + 8 * half);
#pragma unroll
                for (int r = 0; r < 8; r++) {
                    out[(rowb + r) * (size_t)(STEPS * NY) + (size_t)s * NY + ycol] = ya[r];
                }
            }
        }
    }
}

extern "C" void kernel_launch(void* const* d_in, const int* in_sizes, int n_in,
                              void* d_out, int out_size, void* d_ws, size_t ws_size,
                              hipStream_t stream) {
    (void)in_sizes; (void)n_in; (void)out_size; (void)ws_size;
    const float* x   = (const float*)d_in[0];
    const float* W1  = (const float*)d_in[1];
    const float* b1  = (const float*)d_in[2];
    const float* W2  = (const float*)d_in[3];
    const float* b2  = (const float*)d_in[4];
    const float* W3  = (const float*)d_in[5];
    const float* b3  = (const float*)d_in[6];
    const float* Wih = (const float*)d_in[7];
    const float* Whh = (const float*)d_in[8];
    const float* bih = (const float*)d_in[9];
    const float* bhh = (const float*)d_in[10];
    const float* Wo  = (const float*)d_in[11];
    const float* bo  = (const float*)d_in[12];
    float* out = (float*)d_out;

    char* ws = (char*)d_ws;
    float*  h1  = (float*)ws;                                 // 16384*128*4 = 8 MB
    float*  h2  = (float*)(ws + (size_t)8388608);             // 8 MB
    __bf16* h0  = (__bf16*)(ws + (size_t)16777216);           // 16384*256*2 = 8 MB
    __bf16* WgB = (__bf16*)(ws + (size_t)25165824);           // 512 KB
    float*  bg  = (float*) (ws + (size_t)25690112);           // 4 KB
    __bf16* WoB = (__bf16*)(ws + (size_t)25694208);           // 56 KB

    prep_gates<<<1024, 256, 0, stream>>>(Wih, Whh, bih, bhh, Wo, WgB, bg, WoB);
    mlp_layer<false><<<(BATCH * 128) / 256, 256, 0, stream>>>(x,  W1, b1, h1, IN_DIM, 128);
    mlp_layer<false><<<(BATCH * 128) / 256, 256, 0, stream>>>(h1, W2, b2, h2, 128, 128);
    mlp_layer<true ><<<(BATCH * 256) / 256, 256, 0, stream>>>(h2, W3, b3, h0, 128, HID);
    lstm_kernel<<<BATCH / 32, 512, 0, stream>>>(h0, WgB, bg, WoB, bo, out);
}
